// FeatureEngineeringNN_21784074125708
// MI455X (gfx1250) — matrix operand order repaired and checked by reading, not compiled, not run on hardware
//
#include <hip/hip_runtime.h>

typedef __attribute__((ext_vector_type(16))) _Float16 v16h;
typedef __attribute__((ext_vector_type(8)))  _Float16 v8h;
typedef __attribute__((ext_vector_type(8)))  float    v8f;
typedef __attribute__((ext_vector_type(4)))  float    vf4;

#define F_FEAT 310
#define KD     (F_FEAT - 1)              // 309
#define BTOT   2048
#define H      32
#define KCH    32
#define NCHUNK ((KD + KCH - 1) / KCH)    // 10 (last chunk zero-padded)
#define BT     64                        // batch rows per workgroup
#define LSTR   40                        // fp16 LDS row stride (80B: 16B-aligned, bank-spread)
#define OSTR   36                        // f32 out-staging row stride (144B: 16B-aligned)
#define FRAG   512                       // halves per B-fragment (32 lanes x 16)
#define ZOFF   (F_FEAT * BTOT)           // start of zero pad in x16
#define ZPAD   21504                     // > 2048 + 15*4*KD + 64: covers all padded reads

// Build a v16h fragment from two 16-byte loads (ds/global _b128 x2).
__device__ __forceinline__ v16h ld_frag(const _Float16* p, int hi_off) {
    v8h lo = *(const v8h*)(p);
    v8h hi = *(const v8h*)(p + hi_off);
    return __builtin_shufflevector(lo, hi, 0,1,2,3,4,5,6,7,8,9,10,11,12,13,14,15);
}

// ---------------------------------------------------------------------------
// Prep kernels: fp16 x copy (+ zero pad) + W matrices pre-swizzled into the
// WMMA B-fragment register order: idx = ((f*NC + kc)*2 + tile)*512 + lane*16 + h
// where  K = kc*32 + h + ((lane>>4)&1)*16,  n = tile*16 + (lane&15).
// ---------------------------------------------------------------------------
__global__ __launch_bounds__(256) void prep_x16(const float* __restrict__ x,
                                                _Float16* __restrict__ x16,
                                                int n, int total) {
    const int i = blockIdx.x * 256 + threadIdx.x;
    if (i < n)           x16[i] = (_Float16)x[i];
    else if (i < total)  x16[i] = (_Float16)0.f;     // zero pad for OOB lanes
}

__global__ __launch_bounds__(256) void prep_w1t(const float* __restrict__ W1,
                                                _Float16* __restrict__ w1t) {
    const int i = blockIdx.x * 256 + threadIdx.x;
    if (i >= F_FEAT * NCHUNK * 2 * FRAG) return;
    const int h    = i & 15;
    const int lane = (i >> 4) & 31;
    const int tile = (i >> 9) & 1;
    const int kc   = (i >> 10) % NCHUNK;
    const int f    = i / (NCHUNK * 2 * FRAG);
    const int K    = kc * KCH + h + ((lane >> 4) & 1) * 16;
    const int n    = tile * 16 + (lane & 15);
    w1t[i] = (K < KD) ? (_Float16)W1[(f * KD + K) * H + n] : (_Float16)0.f;
}

__global__ __launch_bounds__(256) void prep_w23t(const float* __restrict__ W2,
                                                 const float* __restrict__ W3,
                                                 _Float16* __restrict__ w2t,
                                                 _Float16* __restrict__ w3t) {
    const int i = blockIdx.x * 256 + threadIdx.x;
    if (i >= F_FEAT * 2 * FRAG) return;
    const int h    = i & 15;
    const int lane = (i >> 4) & 31;
    const int tile = (i >> 9) & 1;
    const int f    = i >> 10;
    const int K    = h + ((lane >> 4) & 1) * 16;
    const int n    = tile * 16 + (lane & 15);
    w2t[i] = (_Float16)W2[(f * H + K) * H + n];
    w3t[i] = (_Float16)W3[(f * H + K) * H + n];
}

// ---------------------------------------------------------------------------
// Main kernel (fast path).
// flat_f[t] = x16[t + (t >= f*2048 ? 2048 : 0)]  (linear copy with a hole).
// The hole threshold is outside this block's t-window for ~97% of blocks ->
// uniform-scalar classification folds the shift into the base pointer.
// ---------------------------------------------------------------------------
__global__ __launch_bounds__(128)
void sfp_mlp_wmma(const _Float16* __restrict__ x16,
                  const _Float16* __restrict__ w1t,
                  const _Float16* __restrict__ w2t,
                  const _Float16* __restrict__ w3t,
                  const float* __restrict__ b1, const float* __restrict__ b2,
                  const float* __restrict__ b3, float* __restrict__ out)
{
    __shared__ _Float16 Xs [2][BT][LSTR];   // A chunk (double-buffered): [b][k] fp16
    __shared__ _Float16 Hs [4][16][LSTR];   // per-wave hidden activations [m][k]
    __shared__ float    OutS[BT][OSTR];     // f32 output staging for coalesced stores

    const int tid  = threadIdx.x;
    const int lane = tid & 31;
    const int wave = tid >> 5;
    const int f    = blockIdx.y;
    const int b0   = blockIdx.x * BT;

    const int nr   = lane & 15;             // M-row (A) / N-col (B,C) within 16-tile
    const int hiL  = lane >> 4;
    const int kx   = tid & 31;              // invariant fast index for X staging
    const int cx   = tid >> 5;              // staging row base (steps by 4 per unroll)
    const int hole = f << 11;               // f*2048: skip threshold

    // Block-uniform hole classification (scalar branch, EXEC untouched).
    const bool allhi = hole <= b0 * KD;               // every t >= hole
    const bool alllo = hole >= (b0 + BT) * KD;        // every t <  hole
    const bool uni   = allhi || alllo;
    const _Float16* xb = x16 + (allhi ? BTOT : 0);    // shift folded into base

    _Float16 xr[16];                        // register prefetch of next X chunk

    auto stage_x = [&](int kc) {
        const int kk = kc * KCH + kx;       // k invariant per thread
        const int t0 = (kk < KD) ? (b0 + cx) * KD + kk : ZOFF;  // OOB -> zero pad
        if (uni) {
            const _Float16* p = xb + t0;    // one base, constant instr offsets
            #pragma unroll
            for (int i = 0; i < 16; ++i) xr[i] = p[i * 4 * KD];
        } else {                            // rare straddling block: per-elem select
            int t = t0;
            #pragma unroll
            for (int i = 0; i < 16; ++i) {
                xr[i] = x16[t + (t >= hole ? BTOT : 0)];
                t += 4 * KD;
            }
        }
    };
    auto commit = [&](int buf) {
        #pragma unroll
        for (int i = 0; i < 16; ++i) Xs[buf][cx + 4 * i][kx] = xr[i];
    };

    // Per-wave base into pre-swizzled W1 fragments.
    const _Float16* w1p = w1t + (long)f * (NCHUNK * 2 * FRAG) + lane * 16;

    // ---------------- Layer 1: (64x309) @ (309x32), double-pipelined ---------
    // Fully unrolled: buffer parity, guards and W-frag ping-pong become SSA
    // (no per-iteration register copies, shared load bases + const offsets).
    v16h wf0 = ld_frag(w1p, 8);             // W frags prefetched a chunk ahead
    v16h wf1 = ld_frag(w1p + FRAG, 8);
    stage_x(0);
    commit(0);
    v8f acc0 = {}, acc1 = {};
    #pragma unroll
    for (int kc = 0; kc < NCHUNK; ++kc) {
        __syncthreads();                                   // buf[kc&1] visible
        const bool more = (kc + 1 < NCHUNK);
        if (more) stage_x(kc + 1);                         // X loads fly under WMMA
        v16h nw0 = wf0, nw1 = wf1;
        if (more) {                                        // next W frags in flight
            const _Float16* p = w1p + (kc + 1) * (2 * FRAG);
            nw0 = ld_frag(p, 8);
            nw1 = ld_frag(p + FRAG, 8);
        }
        const v16h a = ld_frag(&Xs[kc & 1][wave * 16 + nr][hiL * 16], 8);
        acc0 = __builtin_amdgcn_wmma_f32_16x16x32_f16(false, a, false, wf0, (short)0, acc0, false, false);
        acc1 = __builtin_amdgcn_wmma_f32_16x16x32_f16(false, a, false, wf1, (short)0, acc1, false, false);
        if (more) commit((kc + 1) & 1);
        wf0 = nw0; wf1 = nw1;
    }

    // Layer-2 W frags early: latency hides under the h1 spill + barrier.
    const _Float16* w2p = w2t + (long)f * (2 * FRAG) + lane * 16;
    const v16h w20 = ld_frag(w2p, 8);
    const v16h w21 = ld_frag(w2p + FRAG, 8);

    // Bias + ReLU, spill h1 to LDS (C layout -> A layout round-trip).
    {
        const float bia0 = b1[f * H + nr];
        const float bia1 = b1[f * H + nr + 16];
        const int mo = hiL * 8;               // C layout: reg r holds M = r (+8 hi)
        #pragma unroll
        for (int r = 0; r < 8; ++r) {
            Hs[wave][mo + r][nr]      = (_Float16)fmaxf(acc0[r] + bia0, 0.f);
            Hs[wave][mo + r][nr + 16] = (_Float16)fmaxf(acc1[r] + bia1, 0.f);
        }
    }
    __syncthreads();

    // ---------------- Layer 2 ----------------
    v8f c20 = {}, c21 = {};
    {
        const v16h a2 = ld_frag(&Hs[wave][nr][hiL * 16], 8);
        c20 = __builtin_amdgcn_wmma_f32_16x16x32_f16(false, a2, false, w20, (short)0, c20, false, false);
        c21 = __builtin_amdgcn_wmma_f32_16x16x32_f16(false, a2, false, w21, (short)0, c21, false, false);
    }
    __syncthreads();

    // Layer-3 W frags early as well.
    const _Float16* w3p = w3t + (long)f * (2 * FRAG) + lane * 16;
    const v16h w30 = ld_frag(w3p, 8);
    const v16h w31 = ld_frag(w3p + FRAG, 8);
    {
        const float bia0 = b2[f * H + nr];
        const float bia1 = b2[f * H + nr + 16];
        const int mo = hiL * 8;
        #pragma unroll
        for (int r = 0; r < 8; ++r) {
            Hs[wave][mo + r][nr]      = (_Float16)fmaxf(c20[r] + bia0, 0.f);
            Hs[wave][mo + r][nr + 16] = (_Float16)fmaxf(c21[r] + bia1, 0.f);
        }
    }
    __syncthreads();

    // ---------------- Layer 3 ----------------
    v8f c30 = {}, c31 = {};
    {
        const v16h a3 = ld_frag(&Hs[wave][nr][hiL * 16], 8);
        c30 = __builtin_amdgcn_wmma_f32_16x16x32_f16(false, a3, false, w30, (short)0, c30, false, false);
        c31 = __builtin_amdgcn_wmma_f32_16x16x32_f16(false, a3, false, w31, (short)0, c31, false, false);
    }

    // Bias + ReLU into f32 staging tile, then fully-coalesced NT b128 stores.
    {
        const float bia0 = b3[f * H + nr];
        const float bia1 = b3[f * H + nr + 16];
        const int mo = hiL * 8;
        #pragma unroll
        for (int r = 0; r < 8; ++r) {
            const int m = wave * 16 + mo + r;
            OutS[m][nr]      = fmaxf(c30[r] + bia0, 0.f);
            OutS[m][nr + 16] = fmaxf(c31[r] + bia1, 0.f);
        }
    }
    __syncthreads();
    {
        const int m  = tid >> 1;
        const int n0 = (tid & 1) * 16;
        float* op = out + ((long)f * BTOT + b0) * H + tid * 16;
        #pragma unroll
        for (int j = 0; j < 4; ++j) {
            const vf4 v = *(const vf4*)&OutS[m][n0 + 4 * j];
            __builtin_nontemporal_store(v, (vf4*)op + j);   // keep 81MB output out of L2
        }
    }
}

// ---------------------------------------------------------------------------
// Fallback (self-contained, no workspace).
// ---------------------------------------------------------------------------
__global__ __launch_bounds__(128)
void sfp_mlp_wmma_fb(const float* __restrict__ x,
                     const float* __restrict__ W1, const float* __restrict__ b1,
                     const float* __restrict__ W2, const float* __restrict__ b2,
                     const float* __restrict__ W3, const float* __restrict__ b3,
                     float* __restrict__ out)
{
    __shared__ _Float16 Xs [2][BT][LSTR];
    __shared__ _Float16 Ws1[2][H][LSTR];
    __shared__ _Float16 W2s[H][LSTR];
    __shared__ _Float16 W3s[H][LSTR];
    __shared__ _Float16 Hs [4][16][LSTR];
    __shared__ float    OutS[BT][OSTR];

    const int tid  = threadIdx.x;
    const int lane = tid & 31;
    const int wave = tid >> 5;
    const int f    = blockIdx.y;
    const int b0   = blockIdx.x * BT;
    const int nr   = lane & 15;
    const int hiL  = lane >> 4;
    const int kx   = tid & 31;
    const int cx   = tid >> 5;
    const int hole = f << 11;

    float xr[16];
    float wr[8];

    auto stage_x = [&](int kc) {
        const int  kk  = kc * KCH + kx;
        const bool ok  = kk < KD;
        int t = (b0 + cx) * KD + (ok ? kk : 0);
        #pragma unroll
        for (int i = 0; i < 16; ++i) {
            const float v = x[t + (t >= hole ? BTOT : 0)];
            xr[i] = ok ? v : 0.f;
            t += 4 * KD;
        }
    };
    auto stage_w = [&](int kc) {
        #pragma unroll
        for (int i = 0; i < 8; ++i) {
            const int  kk  = kc * KCH + cx + 4 * i;
            const bool ok  = kk < KD;
            const int  kkc = ok ? kk : 0;
            const float v  = W1[(f * KD + kkc) * H + kx];
            wr[i] = ok ? v : 0.f;
        }
    };
    auto commit = [&](int buf) {
        #pragma unroll
        for (int i = 0; i < 16; ++i) Xs[buf][cx + 4 * i][kx] = (_Float16)xr[i];
        #pragma unroll
        for (int i = 0; i < 8; ++i)  Ws1[buf][kx][cx + 4 * i] = (_Float16)wr[i];
    };

    #pragma unroll
    for (int i = 0; i < 8; ++i) {
        const int k = cx + 4 * i, n = kx;
        W2s[n][k] = (_Float16)W2[(f * H + k) * H + n];
        W3s[n][k] = (_Float16)W3[(f * H + k) * H + n];
    }

    stage_x(0); stage_w(0);
    commit(0);
    v8f acc0 = {}, acc1 = {};
    for (int kc = 0; kc < NCHUNK; ++kc) {
        __syncthreads();
        if (kc + 1 < NCHUNK) { stage_x(kc + 1); stage_w(kc + 1); }
        const int bsel = kc & 1;
        const v16h a  = ld_frag(&Xs[bsel][wave * 16 + nr][hiL * 16], 8);
        const v16h g0 = ld_frag(&Ws1[bsel][nr][hiL * 16], 8);
        const v16h g1 = ld_frag(&Ws1[bsel][nr + 16][hiL * 16], 8);
        acc0 = __builtin_amdgcn_wmma_f32_16x16x32_f16(false, a, false, g0, (short)0, acc0, false, false);
        acc1 = __builtin_amdgcn_wmma_f32_16x16x32_f16(false, a, false, g1, (short)0, acc1, false, false);
        if (kc + 1 < NCHUNK) commit((kc + 1) & 1);
    }
    {
        const float bia0 = b1[f * H + nr];
        const float bia1 = b1[f * H + nr + 16];
        const int mo = hiL * 8;
        #pragma unroll
        for (int r = 0; r < 8; ++r) {
            Hs[wave][mo + r][nr]      = (_Float16)fmaxf(acc0[r] + bia0, 0.f);
            Hs[wave][mo + r][nr + 16] = (_Float16)fmaxf(acc1[r] + bia1, 0.f);
        }
    }
    __syncthreads();
    v8f c20 = {}, c21 = {};
    {
        const v16h a2  = ld_frag(&Hs[wave][nr][hiL * 16], 8);
        const v16h w20 = ld_frag(&W2s[nr][hiL * 16], 8);
        const v16h w21 = ld_frag(&W2s[nr + 16][hiL * 16], 8);
        c20 = __builtin_amdgcn_wmma_f32_16x16x32_f16(false, a2, false, w20, (short)0, c20, false, false);
        c21 = __builtin_amdgcn_wmma_f32_16x16x32_f16(false, a2, false, w21, (short)0, c21, false, false);
    }
    __syncthreads();
    {
        const float bia0 = b2[f * H + nr];
        const float bia1 = b2[f * H + nr + 16];
        const int mo = hiL * 8;
        #pragma unroll
        for (int r = 0; r < 8; ++r) {
            Hs[wave][mo + r][nr]      = (_Float16)fmaxf(c20[r] + bia0, 0.f);
            Hs[wave][mo + r][nr + 16] = (_Float16)fmaxf(c21[r] + bia1, 0.f);
        }
    }
    __syncthreads();
    v8f c30 = {}, c31 = {};
    {
        const v16h a3  = ld_frag(&Hs[wave][nr][hiL * 16], 8);
        const v16h w30 = ld_frag(&W3s[nr][hiL * 16], 8);
        const v16h w31 = ld_frag(&W3s[nr + 16][hiL * 16], 8);
        c30 = __builtin_amdgcn_wmma_f32_16x16x32_f16(false, a3, false, w30, (short)0, c30, false, false);
        c31 = __builtin_amdgcn_wmma_f32_16x16x32_f16(false, a3, false, w31, (short)0, c31, false, false);
    }
    {
        const float bia0 = b3[f * H + nr];
        const float bia1 = b3[f * H + nr + 16];
        const int mo = hiL * 8;
        #pragma unroll
        for (int r = 0; r < 8; ++r) {
            const int m = wave * 16 + mo + r;
            OutS[m][nr]      = fmaxf(c30[r] + bia0, 0.f);
            OutS[m][nr + 16] = fmaxf(c31[r] + bia1, 0.f);
        }
    }
    __syncthreads();
    {
        const int m  = tid >> 1;
        const int n0 = (tid & 1) * 16;
        float* op = out + ((long)f * BTOT + b0) * H + tid * 16;
        #pragma unroll
        for (int j = 0; j < 4; ++j) {
            const vf4 v = *(const vf4*)&OutS[m][n0 + 4 * j];
            __builtin_nontemporal_store(v, (vf4*)op + j);
        }
    }
}

extern "C" void kernel_launch(void* const* d_in, const int* in_sizes, int n_in,
                              void* d_out, int out_size, void* d_ws, size_t ws_size,
                              hipStream_t stream) {
    (void)in_sizes; (void)n_in; (void)out_size;
    const float* x  = (const float*)d_in[0];
    const float* W1 = (const float*)d_in[1];
    const float* b1 = (const float*)d_in[2];
    const float* W2 = (const float*)d_in[3];
    const float* b2 = (const float*)d_in[4];
    const float* W3 = (const float*)d_in[5];
    const float* b3 = (const float*)d_in[6];
    dim3 grid(BTOT / BT, F_FEAT);

    const size_t nX16 = (size_t)F_FEAT * BTOT;                 // halves
    const size_t nXT  = nX16 + ZPAD;                           // + zero pad
    const size_t nW1T = (size_t)F_FEAT * NCHUNK * 2 * FRAG;    // halves
    const size_t nW2T = (size_t)F_FEAT * 2 * FRAG;             // halves
    const size_t need = (nXT + nW1T + 2 * nW2T) * sizeof(_Float16);

    if (ws_size >= need) {
        _Float16* x16 = (_Float16*)d_ws;
        _Float16* w1t = x16 + nXT;
        _Float16* w2t = w1t + nW1T;
        _Float16* w3t = w2t + nW2T;
        prep_x16 <<<(int)((nXT + 255) / 256), 256, 0, stream>>>(x, x16, (int)nX16, (int)nXT);
        prep_w1t <<<(int)((nW1T + 255) / 256), 256, 0, stream>>>(W1, w1t);
        prep_w23t<<<(int)((nW2T + 255) / 256), 256, 0, stream>>>(W2, W3, w2t, w3t);
        sfp_mlp_wmma<<<grid, 128, 0, stream>>>(x16, w1t, w2t, w3t, b1, b2, b3, (float*)d_out);
    } else {
        sfp_mlp_wmma_fb<<<grid, 128, 0, stream>>>(x, W1, b1, W2, b2, W3, b3, (float*)d_out);
    }
}
